// RWKVLayer_46901042872963
// MI455X (gfx1250) — matrix operand order repaired and checked by reading, not compiled, not run on hardware
//
#include <hip/hip_runtime.h>
#include <hip/hip_bf16.h>
#include <math.h>

typedef __attribute__((ext_vector_type(16))) _Float16 v16h;
typedef __attribute__((ext_vector_type(8)))  _Float16 v8h;
typedef __attribute__((ext_vector_type(8)))  float    v8f;

constexpr int Bc = 8, Tc = 2048, Cc = 1024, Hc = 16, Nc = 64, FFNc = 3584;
constexpr float EPSc = 1e-5f;
constexpr float INV_HSD = 0.125f;   // 1/8

// ---------------------------------------------------------------- utilities

static __device__ inline v16h cat8(v8h lo, v8h hi) {
    v16h r;
#pragma unroll
    for (int i = 0; i < 8; ++i) { r[i] = lo[i]; r[8 + i] = hi[i]; }
    return r;
}

// block-wide dual reduction (sum of a, sum of b) over 256 threads
static __device__ inline void block_sum2(float a, float b, float* r1, float* r2,
                                         float& oa, float& ob) {
    const int t = threadIdx.x;
    r1[t] = a; r2[t] = b;
    __syncthreads();
#pragma unroll
    for (int s = 128; s > 0; s >>= 1) {
        if (t < s) { r1[t] += r1[t + s]; r2[t] += r2[t + s]; }
        __syncthreads();
    }
    oa = r1[0]; ob = r2[0];
    __syncthreads();
}

// ---------------------------------------------------------------- f32 -> f16

__global__ __launch_bounds__(256) void k_cvt_f16(const float* __restrict__ src,
                                                 _Float16* __restrict__ dst,
                                                 size_t n) {
    size_t i = (size_t)blockIdx.x * 256 + threadIdx.x;
    if (i < n) dst[i] = (_Float16)src[i];
}

// ---------------------------------------------------------------- WMMA GEMM
// D[m,n] = sum_c A[m,c] * W[n,c]  (i.e. x @ W.T), A:[M,K] f16, W:[N,K] f16.
// EPI 0: f32 store, optional residual add.  EPI 1: f16 store of relu(x)^2.
// Block: 128 threads (4 waves). Tile 128(M) x 64(N), K step 32.
// LDS tiles are double-buffered and filled with GLOBAL_LOAD_ASYNC_TO_LDS
// (ASYNCcnt path, bypasses VGPRs; INST_OFFSET applies to both addresses).
// Requires M%128==0, N%64==0, K%32==0 (true for all calls here).

constexpr int AROW = 40;           // padded LDS row stride (halves), 80B
constexpr int ASZ  = 128 * AROW;   // halves per A buffer
constexpr int BSZ  = 64 * AROW;    // halves per B buffer

template <int EPI>
__global__ __launch_bounds__(128) void k_gemm(const _Float16* __restrict__ A,
                                              const _Float16* __restrict__ W,
                                              void* __restrict__ Dv,
                                              const float* __restrict__ resid,
                                              int M, int N, int K) {
    __shared__ __attribute__((aligned(16))) _Float16 sA[2 * ASZ];
    __shared__ __attribute__((aligned(16))) _Float16 sB[2 * BSZ];

    const int tid  = threadIdx.x;
    const int wave = tid >> 5;
    const int lane = tid & 31;
    const int bm = blockIdx.y * 128;
    const int bn = blockIdx.x * 64;

    // tile-fill roles: thread fills A row `tid` (64B) and half of B row `tid>>1`
    const int arow = tid;
    const int brow = tid >> 1;
    const int bseg = (tid & 1) * 16;          // halves

    const uint32_t ldsA0 = (uint32_t)(uintptr_t)(sA + arow * AROW);
    const uint32_t ldsB0 = (uint32_t)(uintptr_t)(sB + brow * AROW + bseg);

    const _Float16* gA0 = A + (size_t)(bm + arow) * K;
    const _Float16* gB0 = W + (size_t)(bn + brow) * K + bseg;

    v8f acc[2][4];
#pragma unroll
    for (int s = 0; s < 2; ++s)
#pragma unroll
        for (int nt = 0; nt < 4; ++nt) acc[s][nt] = (v8f)0.0f;

    const int half = lane >> 4;   // 0 or 1
    const int l16  = lane & 15;

    auto issue = [&](int kk, int p) {
        const uint64_t ga = (uint64_t)(uintptr_t)(gA0 + kk);
        const uint64_t gb = (uint64_t)(uintptr_t)(gB0 + kk);
        const uint32_t la = ldsA0 + (uint32_t)(p * (ASZ * 2));
        const uint32_t lb = ldsB0 + (uint32_t)(p * (BSZ * 2));
        asm volatile(
            "global_load_async_to_lds_b128 %0, %1, off\n\t"
            "global_load_async_to_lds_b128 %0, %1, off offset:16\n\t"
            "global_load_async_to_lds_b128 %0, %1, off offset:32\n\t"
            "global_load_async_to_lds_b128 %0, %1, off offset:48"
            :: "v"(la), "v"(ga) : "memory");
        asm volatile(
            "global_load_async_to_lds_b128 %0, %1, off\n\t"
            "global_load_async_to_lds_b128 %0, %1, off offset:16"
            :: "v"(lb), "v"(gb) : "memory");
    };

    issue(0, 0);
    int p = 0;
    for (int kk = 0; kk < K; kk += 32) {
        asm volatile("s_wait_asynccnt 0x0" ::: "memory");
        __syncthreads();                       // buffer p ready block-wide
        if (kk + 32 < K) issue(kk + 32, 1 - p);

        const _Float16* bufA = sA + p * ASZ;
        const _Float16* bufB = sB + p * BSZ;

        v16h af[2];
#pragma unroll
        for (int s = 0; s < 2; ++s) {
            const _Float16* pa = bufA + (32 * wave + 16 * s + l16) * AROW;
            af[s] = cat8(*(const v8h*)(pa + half * 8),
                         *(const v8h*)(pa + 16 + half * 8));
        }
#pragma unroll
        for (int nt = 0; nt < 4; ++nt) {
            const _Float16* pb = bufB + (16 * nt + l16) * AROW + half * 8;
            v16h bf = cat8(*(const v8h*)pb, *(const v8h*)(pb + 16));
            acc[0][nt] = __builtin_amdgcn_wmma_f32_16x16x32_f16(
                false, af[0], false, bf, (short)0, acc[0][nt], false, false);
            acc[1][nt] = __builtin_amdgcn_wmma_f32_16x16x32_f16(
                false, af[1], false, bf, (short)0, acc[1][nt], false, false);
        }
        p ^= 1;
    }

#pragma unroll
    for (int s = 0; s < 2; ++s) {
#pragma unroll
        for (int nt = 0; nt < 4; ++nt) {
#pragma unroll
            for (int v = 0; v < 8; ++v) {
                const int m = bm + 32 * wave + 16 * s + half * 8 + v;
                const int n = bn + 16 * nt + l16;
                const size_t idx = (size_t)m * N + n;
                float val = acc[s][nt][v];
                if (EPI == 0) {
                    if (resid) val += resid[idx];
                    ((float*)Dv)[idx] = val;
                } else {
                    float r = val > 0.0f ? val : 0.0f;
                    ((_Float16*)Dv)[idx] = (_Float16)(r * r);
                }
            }
        }
    }
}

// ------------------------------------------------- fused LN0+LN1+shift+mix

__global__ __launch_bounds__(256) void k_ln01_mix(
    const float* __restrict__ x,
    const float* __restrict__ ln0w, const float* __restrict__ ln0b,
    const float* __restrict__ ln1w, const float* __restrict__ ln1b,
    const float* __restrict__ tmk, const float* __restrict__ tmv,
    const float* __restrict__ tmr, const float* __restrict__ tmg,
    float* __restrict__ x0,
    _Float16* __restrict__ xk, _Float16* __restrict__ xv,
    _Float16* __restrict__ xr, _Float16* __restrict__ xg) {
    __shared__ float r1[256], r2[256];
    const int row = blockIdx.x;
    const int t   = row % Tc;
    const int tid = threadIdx.x;
    const float* cur = x + (size_t)row * Cc;
    const float* prv = x + (size_t)(row - 1) * Cc;   // valid iff t>0

    float xc[4], xp[4];
#pragma unroll
    for (int i = 0; i < 4; ++i) {
        const int c = tid + i * 256;
        xc[i] = cur[c];
        xp[i] = (t > 0) ? prv[c] : 0.0f;
    }

    // ---- current row: ln0 then ln1
    float s, q, mu, var, rs;
    s = xc[0] + xc[1] + xc[2] + xc[3];
    q = xc[0]*xc[0] + xc[1]*xc[1] + xc[2]*xc[2] + xc[3]*xc[3];
    block_sum2(s, q, r1, r2, s, q);
    mu = s / Cc; var = q / Cc - mu * mu; rs = rsqrtf(var + EPSc);
    float ac[4];
#pragma unroll
    for (int i = 0; i < 4; ++i) {
        const int c = tid + i * 256;
        ac[i] = (xc[i] - mu) * rs * ln0w[c] + ln0b[c];
        x0[(size_t)row * Cc + c] = ac[i];
    }
    s = ac[0] + ac[1] + ac[2] + ac[3];
    q = ac[0]*ac[0] + ac[1]*ac[1] + ac[2]*ac[2] + ac[3]*ac[3];
    block_sum2(s, q, r1, r2, s, q);
    mu = s / Cc; var = q / Cc - mu * mu; rs = rsqrtf(var + EPSc);
    float nc[4];
#pragma unroll
    for (int i = 0; i < 4; ++i) {
        const int c = tid + i * 256;
        nc[i] = (ac[i] - mu) * rs * ln1w[c] + ln1b[c];
    }

    // ---- previous row (token shift source): ln0 then ln1; zero at t==0
    float np[4] = {0.f, 0.f, 0.f, 0.f};
    if (t > 0) {   // uniform per block
        s = xp[0] + xp[1] + xp[2] + xp[3];
        q = xp[0]*xp[0] + xp[1]*xp[1] + xp[2]*xp[2] + xp[3]*xp[3];
        block_sum2(s, q, r1, r2, s, q);
        mu = s / Cc; var = q / Cc - mu * mu; rs = rsqrtf(var + EPSc);
        float ap[4];
#pragma unroll
        for (int i = 0; i < 4; ++i) {
            const int c = tid + i * 256;
            ap[i] = (xp[i] - mu) * rs * ln0w[c] + ln0b[c];
        }
        s = ap[0] + ap[1] + ap[2] + ap[3];
        q = ap[0]*ap[0] + ap[1]*ap[1] + ap[2]*ap[2] + ap[3]*ap[3];
        block_sum2(s, q, r1, r2, s, q);
        mu = s / Cc; var = q / Cc - mu * mu; rs = rsqrtf(var + EPSc);
#pragma unroll
        for (int i = 0; i < 4; ++i) {
            const int c = tid + i * 256;
            np[i] = (ap[i] - mu) * rs * ln1w[c] + ln1b[c];
        }
    }

#pragma unroll
    for (int i = 0; i < 4; ++i) {
        const int c = tid + i * 256;
        const size_t idx = (size_t)row * Cc + c;
        float mk = tmk[c], mv = tmv[c], mr = tmr[c], mg = tmg[c];
        xk[idx] = (_Float16)(nc[i] * mk + np[i] * (1.0f - mk));
        xv[idx] = (_Float16)(nc[i] * mv + np[i] * (1.0f - mv));
        xr[idx] = (_Float16)(nc[i] * mr + np[i] * (1.0f - mr));
        xg[idx] = (_Float16)(nc[i] * mg + np[i] * (1.0f - mg));
    }
}

// ------------------------------------------------- fused LN2+shift+mix (FFN)

__global__ __launch_bounds__(256) void k_ln2_mix(
    const float* __restrict__ x,
    const float* __restrict__ lnw, const float* __restrict__ lnb,
    const float* __restrict__ tmk, const float* __restrict__ tmr,
    _Float16* __restrict__ xk, _Float16* __restrict__ xr) {
    __shared__ float r1[256], r2[256];
    const int row = blockIdx.x;
    const int t   = row % Tc;
    const int tid = threadIdx.x;
    const float* cur = x + (size_t)row * Cc;
    const float* prv = x + (size_t)(row - 1) * Cc;

    float xc[4], xp[4];
#pragma unroll
    for (int i = 0; i < 4; ++i) {
        const int c = tid + i * 256;
        xc[i] = cur[c];
        xp[i] = (t > 0) ? prv[c] : 0.0f;
    }
    float s, q, mu, var, rs;
    s = xc[0] + xc[1] + xc[2] + xc[3];
    q = xc[0]*xc[0] + xc[1]*xc[1] + xc[2]*xc[2] + xc[3]*xc[3];
    block_sum2(s, q, r1, r2, s, q);
    mu = s / Cc; var = q / Cc - mu * mu; rs = rsqrtf(var + EPSc);
    float nc[4];
#pragma unroll
    for (int i = 0; i < 4; ++i) {
        const int c = tid + i * 256;
        nc[i] = (xc[i] - mu) * rs * lnw[c] + lnb[c];
    }
    float np[4] = {0.f, 0.f, 0.f, 0.f};
    if (t > 0) {
        s = xp[0] + xp[1] + xp[2] + xp[3];
        q = xp[0]*xp[0] + xp[1]*xp[1] + xp[2]*xp[2] + xp[3]*xp[3];
        block_sum2(s, q, r1, r2, s, q);
        mu = s / Cc; var = q / Cc - mu * mu; rs = rsqrtf(var + EPSc);
#pragma unroll
        for (int i = 0; i < 4; ++i) {
            const int c = tid + i * 256;
            np[i] = (xp[i] - mu) * rs * lnw[c] + lnb[c];
        }
    }
#pragma unroll
    for (int i = 0; i < 4; ++i) {
        const int c = tid + i * 256;
        const size_t idx = (size_t)row * Cc + c;
        float mk = tmk[c], mr = tmr[c];
        xk[idx] = (_Float16)(nc[i] * mk + np[i] * (1.0f - mk));
        xr[idx] = (_Float16)(nc[i] * mr + np[i] * (1.0f - mr));
    }
}

// ------------------------------------------------- wkv5 recurrence
// one block per (b,h), 64 threads; thread j owns state column S[:,j] in regs.

__global__ __launch_bounds__(64) void k_wkv(
    const float* __restrict__ r, const float* __restrict__ k,
    const float* __restrict__ v, const float* __restrict__ decay,
    const float* __restrict__ faaaa, float* __restrict__ y) {
    const int b = blockIdx.x / Hc;
    const int h = blockIdx.x % Hc;
    const int j = threadIdx.x;

    __shared__ float sw[Nc], su[Nc];
    __shared__ float buf[2][3][Nc];      // double-buffered r,k,v rows

    sw[j] = expf(-expf(decay[h * Nc + j]));
    su[j] = faaaa[h * Nc + j];
    __syncthreads();

    float S[Nc];
#pragma unroll
    for (int i = 0; i < Nc; ++i) S[i] = 0.0f;

    for (int t = 0; t < Tc; ++t) {
        const int p = t & 1;
        const size_t idx = ((size_t)(b * Tc + t)) * Cc + h * Nc + j;
        buf[p][0][j] = r[idx];
        buf[p][1][j] = k[idx];
        buf[p][2][j] = v[idx];
        __syncthreads();
        const float vj = buf[p][2][j];
        float acc = 0.0f, cu = 0.0f;
#pragma unroll
        for (int i = 0; i < Nc; ++i) {
            const float ri = buf[p][0][i];
            const float ki = buf[p][1][i];
            acc = fmaf(ri, S[i], acc);
            cu  = fmaf(ri * su[i], ki, cu);
            S[i] = fmaf(sw[i], S[i], ki * vj);
        }
        y[idx] = acc + cu * vj;
    }
}

// ------------------------------------------------- GroupNorm(y/8) * SiLU(g)

__global__ __launch_bounds__(256) void k_gnorm_gate(
    const float* __restrict__ y, const float* __restrict__ g,
    const float* __restrict__ lnxw, const float* __restrict__ lnxb,
    _Float16* __restrict__ out) {
    __shared__ float psum[256], psq[256];
    const int row = blockIdx.x;
    const int tid = threadIdx.x;
    const int c0  = tid * 4;             // 4 consecutive channels, same head
    const int head = tid >> 4;           // 16 threads per head

    float yv[4];
    float s = 0.f, q = 0.f;
#pragma unroll
    for (int i = 0; i < 4; ++i) {
        yv[i] = y[(size_t)row * Cc + c0 + i] * INV_HSD;
        s += yv[i]; q += yv[i] * yv[i];
    }
    psum[tid] = s; psq[tid] = q;
    __syncthreads();
    float hs = 0.f, hq = 0.f;
#pragma unroll
    for (int i = 0; i < 16; ++i) { hs += psum[head * 16 + i]; hq += psq[head * 16 + i]; }
    const float mu = hs / Nc;
    const float var = hq / Nc - mu * mu;
    const float rs = rsqrtf(var + EPSc);
#pragma unroll
    for (int i = 0; i < 4; ++i) {
        const int c = c0 + i;
        const size_t idx = (size_t)row * Cc + c;
        const float gv = g[idx];
        const float silu = gv / (1.0f + expf(-gv));
        const float gn = (yv[i] - mu) * rs * lnxw[c] + lnxb[c];
        out[idx] = (_Float16)(gn * silu);
    }
}

// ------------------------------------------------- final: x + sigmoid(rr)*kv

__global__ __launch_bounds__(256) void k_final(
    const float* __restrict__ x1, const float* __restrict__ rr,
    const float* __restrict__ kv, float* __restrict__ out) {
    const size_t i = (size_t)blockIdx.x * 256 + threadIdx.x;
    const float sg = 1.0f / (1.0f + expf(-rr[i]));
    out[i] = x1[i] + sg * kv[i];
}

// ---------------------------------------------------------------- launcher

extern "C" void kernel_launch(void* const* d_in, const int* in_sizes, int n_in,
                              void* d_out, int out_size, void* d_ws, size_t ws_size,
                              hipStream_t stream) {
    (void)in_sizes; (void)n_in; (void)out_size; (void)ws_size;

    const float* x        = (const float*)d_in[0];
    const float* ln0_w    = (const float*)d_in[1];
    const float* ln0_b    = (const float*)d_in[2];
    const float* ln1_w    = (const float*)d_in[3];
    const float* ln1_b    = (const float*)d_in[4];
    const float* ln2_w    = (const float*)d_in[5];
    const float* ln2_b    = (const float*)d_in[6];
    const float* att_tmk  = (const float*)d_in[7];
    const float* att_tmv  = (const float*)d_in[8];
    const float* att_tmr  = (const float*)d_in[9];
    const float* att_tmg  = (const float*)d_in[10];
    const float* att_decay= (const float*)d_in[11];
    const float* att_faaaa= (const float*)d_in[12];
    const float* att_Wr   = (const float*)d_in[13];
    const float* att_Wk   = (const float*)d_in[14];
    const float* att_Wv   = (const float*)d_in[15];
    const float* att_Wg   = (const float*)d_in[16];
    const float* att_Wo   = (const float*)d_in[17];
    const float* att_lnx_w= (const float*)d_in[18];
    const float* att_lnx_b= (const float*)d_in[19];
    const float* ffn_tmk  = (const float*)d_in[20];
    const float* ffn_tmr  = (const float*)d_in[21];
    const float* ffn_Wk   = (const float*)d_in[22];
    const float* ffn_Wr   = (const float*)d_in[23];
    const float* ffn_Wv   = (const float*)d_in[24];

    const size_t CC  = (size_t)Cc * Cc;
    const size_t FC  = (size_t)FFNc * Cc;
    const size_t BTC = (size_t)Bc * Tc * Cc;

    char* ws = (char*)d_ws;
    size_t off = 0;
    auto alloc = [&](size_t bytes) -> char* {
        char* p = ws + off;
        off = (off + bytes + 255) & ~(size_t)255;
        return p;
    };

    _Float16* wr16  = (_Float16*)alloc(CC * 2);
    _Float16* wk16  = (_Float16*)alloc(CC * 2);
    _Float16* wv16  = (_Float16*)alloc(CC * 2);
    _Float16* wg16  = (_Float16*)alloc(CC * 2);
    _Float16* wo16  = (_Float16*)alloc(CC * 2);
    _Float16* fwk16 = (_Float16*)alloc(FC * 2);
    _Float16* fwr16 = (_Float16*)alloc(CC * 2);
    _Float16* fwv16 = (_Float16*)alloc(FC * 2);
    _Float16* xk16  = (_Float16*)alloc(BTC * 2);   // later reused: ffn xk
    _Float16* xv16  = (_Float16*)alloc(BTC * 2);
    _Float16* xr16  = (_Float16*)alloc(BTC * 2);   // later reused: ffn xr
    _Float16* xg16  = (_Float16*)alloc(BTC * 2);   // later reused: y*g (f16)
    float* x0       = (float*)alloc(BTC * 4);
    float* rbuf     = (float*)alloc(BTC * 4);      // later reused: x1
    float* kvreg    = (float*)alloc(BTC * 8);      // kbuf+vbuf; later kf16 (f16, B*T*FFN)
    float* kbuf     = kvreg;
    float* vbuf     = kvreg + BTC;
    float* gbuf     = (float*)alloc(BTC * 4);      // later reused: kv
    float* ybuf     = (float*)alloc(BTC * 4);      // later reused: rr

    // 1) weight conversion f32 -> f16
    k_cvt_f16<<<(unsigned)(CC / 256), 256, 0, stream>>>(att_Wr, wr16, CC);
    k_cvt_f16<<<(unsigned)(CC / 256), 256, 0, stream>>>(att_Wk, wk16, CC);
    k_cvt_f16<<<(unsigned)(CC / 256), 256, 0, stream>>>(att_Wv, wv16, CC);
    k_cvt_f16<<<(unsigned)(CC / 256), 256, 0, stream>>>(att_Wg, wg16, CC);
    k_cvt_f16<<<(unsigned)(CC / 256), 256, 0, stream>>>(att_Wo, wo16, CC);
    k_cvt_f16<<<(unsigned)(FC / 256), 256, 0, stream>>>(ffn_Wk, fwk16, FC);
    k_cvt_f16<<<(unsigned)(CC / 256), 256, 0, stream>>>(ffn_Wr, fwr16, CC);
    k_cvt_f16<<<(unsigned)(FC / 256), 256, 0, stream>>>(ffn_Wv, fwv16, FC);

    const int M = Bc * Tc;   // 16384

    // 2) LN0 + LN1 + token shift + mixes
    k_ln01_mix<<<M, 256, 0, stream>>>(x, ln0_w, ln0_b, ln1_w, ln1_b,
                                      att_tmk, att_tmv, att_tmr, att_tmg,
                                      x0, xk16, xv16, xr16, xg16);

    // 3) projections r,k,v,g  (WMMA GEMMs)
    dim3 gcc(Cc / 64, M / 128);
    k_gemm<0><<<gcc, 128, 0, stream>>>(xr16, wr16, (void*)rbuf, nullptr, M, Cc, Cc);
    k_gemm<0><<<gcc, 128, 0, stream>>>(xk16, wk16, (void*)kbuf, nullptr, M, Cc, Cc);
    k_gemm<0><<<gcc, 128, 0, stream>>>(xv16, wv16, (void*)vbuf, nullptr, M, Cc, Cc);
    k_gemm<0><<<gcc, 128, 0, stream>>>(xg16, wg16, (void*)gbuf, nullptr, M, Cc, Cc);

    // 4) wkv5 recurrence
    k_wkv<<<Bc * Hc, Nc, 0, stream>>>(rbuf, kbuf, vbuf, att_decay, att_faaaa, ybuf);

    // 5) GroupNorm(y/8) * SiLU(g) -> f16 (into xg16 slot)
    k_gnorm_gate<<<M, 256, 0, stream>>>(ybuf, gbuf, att_lnx_w, att_lnx_b, xg16);

    // 6) output projection with residual: x1 = x0 + (y*g) @ Wo.T  (into rbuf)
    float* x1 = rbuf;
    k_gemm<0><<<gcc, 128, 0, stream>>>(xg16, wo16, (void*)x1, x0, M, Cc, Cc);

    // 7) LN2 + shift + mix (into xk16/xr16 slots)
    k_ln2_mix<<<M, 256, 0, stream>>>(x1, ln2_w, ln2_b, ffn_tmk, ffn_tmr, xk16, xr16);

    // 8) kf = relu(xk @ Wk.T)^2 -> f16 (into kbuf+vbuf region)
    _Float16* kf16 = (_Float16*)kvreg;
    dim3 gfn(FFNc / 64, M / 128);
    k_gemm<1><<<gfn, 128, 0, stream>>>(xk16, fwk16, (void*)kf16, nullptr, M, FFNc, Cc);

    // 9) kv = kf @ Wv.T (into gbuf)
    float* kv = gbuf;
    k_gemm<0><<<gcc, 128, 0, stream>>>(kf16, fwv16, (void*)kv, nullptr, M, Cc, FFNc);

    // 10) rr = xr @ Wr.T (into ybuf)
    float* rr = ybuf;
    k_gemm<0><<<gcc, 128, 0, stream>>>(xr16, fwr16, (void*)rr, nullptr, M, Cc, Cc);

    // 11) out = x1 + sigmoid(rr) * kv
    k_final<<<(unsigned)(BTC / 256), 256, 0, stream>>>(x1, rr, kv, (float*)d_out);
}
